// OpticsAugment_51281909514340
// MI455X (gfx1250) — matrix operand order repaired and checked by reading, not compiled, not run on hardware
//
#include <hip/hip_runtime.h>
#include <hip/hip_bf16.h>
#include <stdint.h>

// ---------------- CDNA5 vector types ----------------
typedef __bf16 v8bf  __attribute__((ext_vector_type(8)));
typedef __bf16 v16bf __attribute__((ext_vector_type(16)));
typedef float  v8f   __attribute__((ext_vector_type(8)));
typedef unsigned int v4u __attribute__((ext_vector_type(4)));
typedef int v4i __attribute__((ext_vector_type(4)));
typedef int v8i __attribute__((ext_vector_type(8)));

#define BSZ   64
#define CSZ   3
#define HSZ   224
#define WSZ   224
#define KSZ   37
#define PADR  18
#define PADH  260           // 224 + 2*18
#define PADW  272           // 224 + 2*18 + 12 slack (16B-aligned rows, tile K=64 fits)
#define PLANES      (BSZ*CSZ)
#define PLANE_PAD   (PADH*PADW)
#define FRAG_ELEMS  512     // 32 lanes * 16 bf16 per WMMA B fragment
#define FRAGS_PC    (KSZ*2) // 74 fragments per (param,channel)
#define NTILEX      14      // 224/16
#define OUT_PIXELS  (BSZ*CSZ*HSZ*WSZ)
#define SLAB_ELEMS  (52*64) // one A-tile: 52 rows x 64 bf16

__device__ __forceinline__ unsigned short f2bf_rne(float f) {
  unsigned u = __float_as_uint(f);
  return (unsigned short)((u + 0x7FFFu + ((u >> 16) & 1u)) >> 16);
}

// ---------- kernel 1a: reflect-padded bf16 image (4 cols / thread, 8B stores) ----------
__global__ void prep_pad(const float* __restrict__ batch,
                         unsigned short* __restrict__ pad) {
  int idx = blockIdx.x * blockDim.x + threadIdx.x;
  const int QW = PADW / 4;                       // 68 quads per row
  const int total = PLANES * PADH * QW;
  if (idx >= total) return;
  int plane = idx / (PADH * QW);
  int rem   = idx - plane * (PADH * QW);
  int r  = rem / QW;
  int cq = (rem - r * QW) * 4;
  int rs = r - PADR; rs = rs < 0 ? -rs : rs; rs = rs > 223 ? 446 - rs : rs;
  const float* src = batch + (size_t)plane * (HSZ * WSZ) + rs * WSZ;
  unsigned v01 = 0, v23 = 0;
  #pragma unroll
  for (int u = 0; u < 4; ++u) {
    int c = cq + u;
    unsigned short v = 0;
    if (c < HSZ + 2 * PADR) {
      int cs = c - PADR; cs = cs < 0 ? -cs : cs; cs = cs > 223 ? 446 - cs : cs;
      v = f2bf_rne(src[cs]);
    }
    if (u < 2) v01 |= (unsigned)v << (16 * u);
    else       v23 |= (unsigned)v << (16 * (u - 2));
  }
  *(uint2*)(pad + (size_t)plane * PLANE_PAD + r * PADW + cq) = make_uint2(v01, v23);
}

// ---------- kernel 1b: Toeplitz B fragments in WMMA lane layout ----------
// Fragment element (lane, t): N = lane&15, K = chunk*32 + (lane>>4)*16 + t,
// value = k_i[K - N] if 0 <= K-N < 37 else 0.  (B 32x16 bf16, contiguous K order, same as the A loads)
__global__ void prep_toep(const float* __restrict__ psf,
                          unsigned short* __restrict__ toep) {
  int idx = blockIdx.x * blockDim.x + threadIdx.x;
  const int total = 8 * 3 * KSZ * 2 * FRAG_ELEMS;
  if (idx >= total) return;
  int t     = idx & 15;
  int lane  = (idx >> 4) & 31;
  int chunk = (idx >> 9) & 1;
  int rest  = idx >> 10;            // ((p*3+c)*37 + i)
  int i  = rest % KSZ;
  int pc = rest / KSZ;
  int c = pc % 3, p = pc / 3;
  int n  = lane & 15;
  int kk = chunk * 32 + ((lane >> 4) << 4) + t;
  int j  = kk - n;
  unsigned short v = 0;
  if (j >= 0 && j < KSZ) {
    // psf_stack[p, SEVERITY-1=2, 0, c, i, j] ; shape (8,5,1,3,37,37)
    v = f2bf_rne(psf[((((p * 5 + 2) * 1 + 0) * 3 + c) * KSZ + i) * KSZ + j]);
  }
  toep[idx] = v;
}

// ---------- kernel 2: Toeplitz-WMMA depthwise conv + lerp, double-buffered TDM ----------
__global__ void __launch_bounds__(128)
conv_wmma(const float* __restrict__ batch,
          const unsigned short* __restrict__ pad,
          const unsigned short* __restrict__ toep,
          const int* __restrict__ params,
          const float* __restrict__ weights,
          float* __restrict__ out) {
  extern __shared__ unsigned short smem[];
  unsigned short* BshU    = smem;                           // 37888 ushorts
  unsigned short* AshBase = smem + FRAGS_PC * FRAG_ELEMS;   // 8 slabs of 3328

  int bid   = blockIdx.x;
  int plane = bid / NTILEX;          // 0..191  (b*3 + channel)
  int ty    = bid - plane * NTILEX;  // 0..13
  int b  = plane / 3, ch = plane - b * 3;
  int p  = params[b];
  float w = weights[b];

  // Stage this plane's 74 Toeplitz fragments (75776 B) into LDS once.
  {
    const uint4* src = (const uint4*)(toep + (size_t)(p * 3 + ch) * FRAGS_PC * FRAG_ELEMS);
    uint4* dst = (uint4*)BshU;
    const int n16 = FRAGS_PC * FRAG_ELEMS * 2 / 16;  // 4736
    for (int k = threadIdx.x; k < n16; k += blockDim.x) dst[k] = src[k];
  }
  __syncthreads();

  int wave = threadIdx.x >> 5;
  int lane = threadIdx.x & 31;
  int m    = lane & 15;
  int hi8  = (lane >> 4) << 4;       // A-lane K sub-offset (0 or 16), contiguous K order
  unsigned short* slab0 = AshBase + (wave * 2 + 0) * SLAB_ELEMS;
  unsigned short* slab1 = AshBase + (wave * 2 + 1) * SLAB_ELEMS;
  int nk = (NTILEX - wave + 3) / 4;  // tiles handled by this wave (3 or 4)

  // TDM issue for tile tx into slab `buf` (descriptor values wave-uniform).
  auto tdm_issue = [&](int tx, unsigned short* slab) {
    unsigned long long gaddr = (unsigned long long)(uintptr_t)pad +
        (((unsigned long long)plane * PLANE_PAD) +
         (unsigned long long)(ty * 16) * PADW + (unsigned)(tx * 16)) * 2ull;
    unsigned ldsA = (unsigned)(uintptr_t)(void*)slab;   // LDS byte offset (addr[31:0])
    v4u g0;
    g0.x = 1u;                                          // count=1 valid descriptor
    g0.y = ldsA;                                        // lds_addr
    g0.z = (unsigned)(gaddr & 0xFFFFFFFFull);           // global_addr lo
    g0.w = (unsigned)((gaddr >> 32) & 0x1FFFFFFull) | (2u << 30); // addr hi | type=2
    v8i g1;
    g1[0] = 0x10000;                 // data_size=1 (2 bytes), workgroup_mask=0
    g1[1] = (int)(PADW << 16);       // tensor_dim0 lo16 @ bits[63:48]
    g1[2] = (int)(PADH << 16);       // tensor_dim1 lo16 @ bits[95:80]
    g1[3] = (int)(64   << 16);       // tile_dim0 = 64 @ bits[127:112]
    g1[4] = 52;                      // tile_dim1 = 52, tile_dim2 = 0
    g1[5] = PADW;                    // tensor_dim0_stride lo32
    g1[6] = 0;
    g1[7] = 0;
    v4i gz4 = {0, 0, 0, 0};
    v8i gz8 = {0, 0, 0, 0, 0, 0, 0, 0};
    // 6-arg form (clang-23 / therock-10.0 headers): (g0, g1, g2, g3, g4, cpol)
    __builtin_amdgcn_tensor_load_to_lds(g0, g1, gz4, gz4, gz8, 0);
  };

  tdm_issue(wave, slab0);            // prologue: first tile in flight

  for (int k = 0; k < nk; ++k) {
    int tx = wave + 4 * k;
    unsigned short* cur = (k & 1) ? slab1 : slab0;
    if (k + 1 < nk) {
      tdm_issue(tx + 4, (k & 1) ? slab0 : slab1);  // prefetch next tile
      __builtin_amdgcn_s_wait_tensorcnt(1);        // older (current) load done
    } else {
      __builtin_amdgcn_s_wait_tensorcnt(0);
    }
    asm volatile("" ::: "memory");   // LDS now holds current tile; block reordering

    v8f acc = {0.f, 0.f, 0.f, 0.f, 0.f, 0.f, 0.f, 0.f};
    for (int i = 0; i < KSZ; ++i) {
      const unsigned short* row = cur + (m + i) * 64;
      const unsigned short* fr0 = BshU + (i * 2 + 0) * FRAG_ELEMS + lane * 16;
      const unsigned short* fr1 = fr0 + FRAG_ELEMS;
      // K-chunk 0 (K = 0..31)
      v8bf a0l = *(const v8bf*)(row + hi8);
      v8bf a0h = *(const v8bf*)(row + 8 + hi8);
      v8bf b0l = *(const v8bf*)(fr0);
      v8bf b0h = *(const v8bf*)(fr0 + 8);
      v16bf a0 = __builtin_shufflevector(a0l, a0h, 0,1,2,3,4,5,6,7,8,9,10,11,12,13,14,15);
      v16bf b0 = __builtin_shufflevector(b0l, b0h, 0,1,2,3,4,5,6,7,8,9,10,11,12,13,14,15);
      acc = __builtin_amdgcn_wmma_f32_16x16x32_bf16(false, a0, false, b0,
                                                    (short)0, acc, false, false);
      // K-chunk 1 (K = 32..63)
      v8bf a1l = *(const v8bf*)(row + 32 + hi8);
      v8bf a1h = *(const v8bf*)(row + 40 + hi8);
      v8bf b1l = *(const v8bf*)(fr1);
      v8bf b1h = *(const v8bf*)(fr1 + 8);
      v16bf a1 = __builtin_shufflevector(a1l, a1h, 0,1,2,3,4,5,6,7,8,9,10,11,12,13,14,15);
      v16bf b1 = __builtin_shufflevector(b1l, b1h, 0,1,2,3,4,5,6,7,8,9,10,11,12,13,14,15);
      acc = __builtin_amdgcn_wmma_f32_16x16x32_bf16(false, a1, false, b1,
                                                    (short)0, acc, false, false);
    }

    // Epilogue: C layout (f32 16x16): VGPR r -> M=r (lanes 0-15), M=8+r (lanes 16-31)
    int x     = tx * 16 + (lane & 15);
    int ybase = ty * 16 + ((lane >> 4) << 3);
    const float* op = batch + (size_t)plane * (HSZ * WSZ);
    float*       dp = out   + (size_t)plane * (HSZ * WSZ);
    #pragma unroll
    for (int r = 0; r < 8; ++r) {
      int y = ybase + r;
      float orig = op[y * WSZ + x];
      dp[y * WSZ + x] = (1.0f - w) * orig + w * acc[r];
    }
  }
}

// ---------- kernel 3: tuple tail (targets passthrough) ----------
__global__ void copy_targets(const int* __restrict__ t, int* __restrict__ dst) {
  int i = threadIdx.x;
  if (i < BSZ) dst[i] = t[i];
}

extern "C" void kernel_launch(void* const* d_in, const int* in_sizes, int n_in,
                              void* d_out, int out_size, void* d_ws, size_t ws_size,
                              hipStream_t stream) {
  (void)in_sizes; (void)n_in; (void)out_size; (void)ws_size;
  const float* batch   = (const float*)d_in[0];
  const int*   targets = (const int*)d_in[1];
  const float* psf     = (const float*)d_in[2];
  const int*   params  = (const int*)d_in[3];
  const float* weights = (const float*)d_in[4];
  float* out = (float*)d_out;

  unsigned short* pad  = (unsigned short*)d_ws;                       // 27.2 MB
  unsigned short* toep = pad + (size_t)PLANES * PLANE_PAD;            // +1.8 MB

  {
    int total = PLANES * PADH * (PADW / 4);
    prep_pad<<<(total + 255) / 256, 256, 0, stream>>>(batch, pad);
  }
  {
    int total = 8 * 3 * KSZ * 2 * FRAG_ELEMS;
    prep_toep<<<(total + 255) / 256, 256, 0, stream>>>(psf, toep);
  }
  {
    size_t lds = (size_t)(FRAGS_PC * FRAG_ELEMS + 4 * 2 * SLAB_ELEMS) * sizeof(unsigned short); // 129024 B
    conv_wmma<<<PLANES * NTILEX, 128, lds, stream>>>(batch, pad, toep, params, weights, out);
  }
  copy_targets<<<1, 64, 0, stream>>>(targets, (int*)d_out + OUT_PIXELS);
}
